// BaselineModel_29334626632569
// MI455X (gfx1250) — matrix operand order repaired and checked by reading, not compiled, not run on hardware
//
#include <hip/hip_runtime.h>
#include <hip/hip_bf16.h>

// ---------------- problem dims (fixed by reference) ----------------
#define T_TOK  16384   // B*S = 8*2048
#define DMODEL 512
#define IDIM   1024
#define NEXP   8
#define SHDIM  2048    // 2*I shared-expert width
// tile config
#define BM 128
#define BN 64
#define BK 32

typedef __attribute__((ext_vector_type(16))) __bf16       bf16x16;
typedef __attribute__((ext_vector_type(8)))  float        f32x8;
typedef __attribute__((ext_vector_type(4)))  float        f32x4;
typedef __attribute__((ext_vector_type(4)))  unsigned int u32x4;

union Frag { bf16x16 v; u32x4 q[2]; };

__device__ __forceinline__ unsigned short f2bf(float f) {
  unsigned int u = __float_as_uint(f);
  u += 0x7FFFu + ((u >> 16) & 1u);          // round-to-nearest-even
  return (unsigned short)(u >> 16);
}
__device__ __forceinline__ float siluf(float v) { return v / (1.0f + __expf(-v)); }

// ---- CDNA5 async global->LDS copy (ASYNCcnt-tracked) ----
__device__ __forceinline__ void async_g2l_b128(unsigned lds_byte_off, const void* gsrc) {
  unsigned long long a = (unsigned long long)(size_t)gsrc;
  asm volatile("global_load_async_to_lds_b128 %0, %1, off"
               :: "v"(lds_byte_off), "v"(a) : "memory");
}
__device__ __forceinline__ void wait_async0() {
  asm volatile("s_wait_asynccnt 0x0" ::: "memory");
}

// ---------------------------------------------------------------------------
// Gate: sigmoid scores, top-2, combine weights, per-expert token lists + stats.
// ---------------------------------------------------------------------------
__global__ __launch_bounds__(256)
void gate_topk(const float* __restrict__ X, const float* __restrict__ GW,
               float* __restrict__ CW, float* __restrict__ stats,
               int* __restrict__ icnt, int* __restrict__ idxlist) {
  const int t    = blockIdx.x * 8 + (threadIdx.x >> 5);
  const int lane = threadIdx.x & 31;
  float acc[NEXP];
#pragma unroll
  for (int e = 0; e < NEXP; ++e) acc[e] = 0.0f;
  for (int i = lane; i < DMODEL; i += 32) {
    float xv = X[(size_t)t * DMODEL + i];
#pragma unroll
    for (int e = 0; e < NEXP; ++e) acc[e] += xv * GW[e * DMODEL + i];
  }
#pragma unroll
  for (int e = 0; e < NEXP; ++e)
    for (int off = 16; off; off >>= 1) acc[e] += __shfl_xor(acc[e], off, 32);
  if (lane == 0) {
    float s[NEXP];
#pragma unroll
    for (int e = 0; e < NEXP; ++e) s[e] = 1.0f / (1.0f + __expf(-acc[e]));
    int i1 = 0;
    for (int e = 1; e < NEXP; ++e) if (s[e] > s[i1]) i1 = e;
    int i2 = (i1 == 0) ? 1 : 0;
    for (int e = 0; e < NEXP; ++e) if (e != i1 && s[e] > s[i2]) i2 = e;
    float sum = s[i1] + s[i2];
    float wt1 = s[i1] / sum, wt2 = s[i2] / sum;  // ROUTE_SCALE == 1.0
#pragma unroll
    for (int e = 0; e < NEXP; ++e) CW[t * NEXP + e] = 0.0f;
    CW[t * NEXP + i1] = wt1;
    CW[t * NEXP + i2] = wt2;
    // sparse routing lists (order nondeterministic; output values are not)
    int p1 = atomicAdd(&icnt[i1], 1);
    idxlist[i1 * T_TOK + p1] = t;
    int p2 = atomicAdd(&icnt[i2], 1);
    idxlist[i2 * T_TOK + p2] = t;
    // aux-loss stats
    atomicAdd(&stats[i1], 1.0f);
    atomicAdd(&stats[i2], 1.0f);
    atomicAdd(&stats[8 + i1], wt1);
    atomicAdd(&stats[8 + i2], wt2);
  }
}

__global__ void zero_stats(float* stats, int* icnt) {
  if (threadIdx.x < 16) stats[threadIdx.x] = 0.0f;
  if (threadIdx.x < NEXP) icnt[threadIdx.x] = 0;
}

// pad each expert list to a multiple of BM with token 0; record padded count
__global__ void pad_lists(const int* __restrict__ icnt, int* __restrict__ icntPad,
                          int* __restrict__ idxlist) {
  const int e = blockIdx.x;
  const int c = icnt[e];
  const int cp = ((c + BM - 1) / BM) * BM;
  if (threadIdx.x == 0) icntPad[e] = cp;
  for (int i = c + threadIdx.x; i < cp; i += blockDim.x) idxlist[e * T_TOK + i] = 0;
}

__global__ void aux_loss(const float* __restrict__ stats, float* __restrict__ outL) {
  if (threadIdx.x == 0) {
    float L = 0.0f;
    for (int e = 0; e < NEXP; ++e) {
      float f = (float)NEXP * stats[e] / (float)(2 * T_TOK);
      float P = stats[8 + e] / (float)T_TOK;
      L += f * P;
    }
    *outL = L;
  }
}

// ---------------------------------------------------------------------------
// Precision prep: convert-once so all GEMM staging becomes pure async copies.
// ---------------------------------------------------------------------------
__global__ __launch_bounds__(256)
void to_bf16(const float* __restrict__ src, unsigned short* __restrict__ dst) {
  size_t i = ((size_t)blockIdx.x * 256 + threadIdx.x) * 4;
  f32x4 f = *(const f32x4*)(src + i);
  dst[i + 0] = f2bf(f.x); dst[i + 1] = f2bf(f.y);
  dst[i + 2] = f2bf(f.z); dst[i + 3] = f2bf(f.w);
}

// src f32 [K][N] -> dst bf16 [N][K]; batched over blockIdx.z
__global__ __launch_bounds__(256)
void transpose_to_bf16(const float* __restrict__ src, unsigned short* __restrict__ dst,
                       int K, int N, long srcStride, long dstStride) {
  src += (size_t)blockIdx.z * srcStride;
  dst += (size_t)blockIdx.z * dstStride;
  const int n0 = blockIdx.x * 32, k0 = blockIdx.y * 32;
  const int k = threadIdx.x >> 3;
  const int n = (threadIdx.x & 7) * 4;
  f32x4 f = *(const f32x4*)(src + (size_t)(k0 + k) * N + n0 + n);
  dst[(size_t)(n0 + n + 0) * K + k0 + k] = f2bf(f.x);
  dst[(size_t)(n0 + n + 1) * K + k0 + k] = f2bf(f.y);
  dst[(size_t)(n0 + n + 2) * K + k0 + k] = f2bf(f.z);
  dst[(size_t)(n0 + n + 3) * K + k0 + k] = f2bf(f.w);
}

// ---------------------------------------------------------------------------
// Dual GEMM + SwiGLU:  H[pos,n] = bf16( cw(tok)*silu(x@W1+b1)*(x@W3+b3) )
// Dense (idx==null): pos == tok.  Sparse: tok = idx[pos] row-gather; the
// gathered row's K-slice is still a contiguous 16B chunk -> async staging.
// ---------------------------------------------------------------------------
__global__ __launch_bounds__(256)
void dual_swiglu(const unsigned short* __restrict__ Xb,
                 const unsigned short* __restrict__ W1t,
                 const unsigned short* __restrict__ W3t,
                 const float* __restrict__ B1, const float* __restrict__ B3,
                 const float* __restrict__ CW, int eidx,
                 const int* __restrict__ idx, const int* __restrict__ pcntPad,
                 unsigned short* __restrict__ H, int N) {
  const int row0 = blockIdx.x * BM;
  if (idx && row0 >= *pcntPad) return;   // uniform early-exit (sparse tail)

  // per buffer: X 4096 ushorts (8KB) + W1 2048 (4KB) + W3 2048 (4KB)
  __shared__ unsigned short smem[2 * 8192];

  const int tid  = threadIdx.x;
  const int lane = tid & 31;
  const int wid  = tid >> 5;
  const int wm   = wid >> 1;
  const int wn   = wid & 1;
  const int col0 = blockIdx.y * BN;
  const int hl   = lane >> 4;
  const int l16  = lane & 15;

  const unsigned sbase = (unsigned)(size_t)(void*)smem;
  const int xr = tid >> 1, xh = tid & 1;
  const int wn_row = tid >> 2, wpart = (tid & 3) * 8;

  const int tokr = idx ? idx[row0 + xr] : (row0 + xr);   // hoisted row gather
  const unsigned short* xrow = Xb + (size_t)tokr * DMODEL + xh * 16;

  auto issue_tile = [&](int k0, int b) {
    const unsigned base = sbase + b * 16384u;
    unsigned xo = base + (unsigned)(xr * 64 + xh * 32);
    async_g2l_b128(xo, xrow + k0);
    async_g2l_b128(xo + 16u, xrow + k0 + 8);
    unsigned w1o = base + 8192u + (unsigned)(wn_row * 64 + wpart * 2);
    async_g2l_b128(w1o, W1t + (size_t)(col0 + wn_row) * DMODEL + k0 + wpart);
    unsigned w3o = base + 12288u + (unsigned)(wn_row * 64 + wpart * 2);
    async_g2l_b128(w3o, W3t + (size_t)(col0 + wn_row) * DMODEL + k0 + wpart);
  };

  f32x8 acc1[2][2], acc3[2][2];
  const f32x8 zero = {0.f, 0.f, 0.f, 0.f, 0.f, 0.f, 0.f, 0.f};
#pragma unroll
  for (int i = 0; i < 2; ++i)
#pragma unroll
    for (int j = 0; j < 2; ++j) { acc1[i][j] = zero; acc3[i][j] = zero; }

  issue_tile(0, 0);
  const int nk = DMODEL / BK;  // 16
  for (int it = 0; it < nk; ++it) {
    const int b = it & 1;
    wait_async0();
    __syncthreads();
    if (it + 1 < nk) issue_tile((it + 1) * BK, b ^ 1);  // overlaps with compute

    const unsigned short* xs  = smem + b * 8192;
    const unsigned short* w1s = xs + 4096;
    const unsigned short* w3s = xs + 6144;

    Frag a[2], fb1[2], fb3[2];
#pragma unroll
    for (int mi = 0; mi < 2; ++mi) {
      const unsigned short* ap = &xs[(wm * 32 + mi * 16 + l16) * BK];
      a[mi].q[0] = *(const u32x4*)(ap + hl * 16);
      a[mi].q[1] = *(const u32x4*)(ap + hl * 16 + 8);
    }
#pragma unroll
    for (int ni = 0; ni < 2; ++ni) {
      const unsigned short* b1p = &w1s[(wn * 32 + ni * 16 + l16) * BK + hl * 16];
      fb1[ni].q[0] = *(const u32x4*)(b1p);
      fb1[ni].q[1] = *(const u32x4*)(b1p + 8);
      const unsigned short* b3p = &w3s[(wn * 32 + ni * 16 + l16) * BK + hl * 16];
      fb3[ni].q[0] = *(const u32x4*)(b3p);
      fb3[ni].q[1] = *(const u32x4*)(b3p + 8);
    }
#pragma unroll
    for (int mi = 0; mi < 2; ++mi)
#pragma unroll
      for (int ni = 0; ni < 2; ++ni) {
        acc1[mi][ni] = __builtin_amdgcn_wmma_f32_16x16x32_bf16(
            false, a[mi].v, false, fb1[ni].v, (short)0, acc1[mi][ni], false, false);
        acc3[mi][ni] = __builtin_amdgcn_wmma_f32_16x16x32_bf16(
            false, a[mi].v, false, fb3[ni].v, (short)0, acc3[mi][ni], false, false);
      }
    __syncthreads();  // all reads of buffer b done before it is re-filled
  }

#pragma unroll
  for (int mi = 0; mi < 2; ++mi) {
#pragma unroll
    for (int ni = 0; ni < 2; ++ni) {
      const int colN = col0 + wn * 32 + ni * 16 + l16;
      const float bb1 = B1 ? B1[colN] : 0.0f;
      const float bb3 = B3 ? B3[colN] : 0.0f;
#pragma unroll
      for (int r = 0; r < 8; ++r) {
        const int pos = row0 + wm * 32 + mi * 16 + hl * 8 + r;
        float h = siluf(acc1[mi][ni][r] + bb1) * (acc3[mi][ni][r] + bb3);
        if (CW) h *= CW[idx[pos] * NEXP + eidx];  // fold routing weight (sparse)
        H[(size_t)pos * N + colN] = f2bf(h);
      }
    }
  }
}

// ---------------------------------------------------------------------------
// Dense (idx==null): Out[pos,:] = A@Wt + bias.
// Sparse: Out[idx[pos],:] += A@Wt, masked to pos < *pcnt (skip padding rows).
// ---------------------------------------------------------------------------
__global__ __launch_bounds__(256)
void gemm_bf16_acc(const unsigned short* __restrict__ A,
                   const unsigned short* __restrict__ Wt,
                   const float* __restrict__ bias,
                   float* __restrict__ Out, int K,
                   const int* __restrict__ idx,
                   const int* __restrict__ pcnt, const int* __restrict__ pcntPad) {
  const int row0 = blockIdx.x * BM;
  if (idx && row0 >= *pcntPad) return;
  const int cnt = idx ? *pcnt : 0;

  __shared__ unsigned short smem[2 * 6144];

  const int tid  = threadIdx.x;
  const int lane = tid & 31;
  const int wid  = tid >> 5;
  const int wm   = wid >> 1;
  const int wn   = wid & 1;
  const int col0 = blockIdx.y * BN;
  const int hl   = lane >> 4;
  const int l16  = lane & 15;

  const unsigned sbase = (unsigned)(size_t)(void*)smem;
  const int xr = tid >> 1, xh = tid & 1;
  const int wn_row = tid >> 2, wpart = (tid & 3) * 8;

  const unsigned short* arow = A + (size_t)(row0 + xr) * K + xh * 16;

  auto issue_tile = [&](int k0, int b) {
    const unsigned base = sbase + b * 12288u;
    unsigned ao = base + (unsigned)(xr * 64 + xh * 32);
    async_g2l_b128(ao, arow + k0);
    async_g2l_b128(ao + 16u, arow + k0 + 8);
    unsigned wo = base + 8192u + (unsigned)(wn_row * 64 + wpart * 2);
    async_g2l_b128(wo, Wt + (size_t)(col0 + wn_row) * K + k0 + wpart);
  };

  f32x8 acc[2][2];
  const f32x8 zero = {0.f, 0.f, 0.f, 0.f, 0.f, 0.f, 0.f, 0.f};
#pragma unroll
  for (int i = 0; i < 2; ++i)
#pragma unroll
    for (int j = 0; j < 2; ++j) acc[i][j] = zero;

  issue_tile(0, 0);
  const int nk = K / BK;
  for (int it = 0; it < nk; ++it) {
    const int b = it & 1;
    wait_async0();
    __syncthreads();
    if (it + 1 < nk) issue_tile((it + 1) * BK, b ^ 1);

    const unsigned short* as = smem + b * 6144;
    const unsigned short* bs = as + 4096;

    Frag a[2], fb[2];
#pragma unroll
    for (int mi = 0; mi < 2; ++mi) {
      const unsigned short* ap = &as[(wm * 32 + mi * 16 + l16) * BK];
      a[mi].q[0] = *(const u32x4*)(ap + hl * 16);
      a[mi].q[1] = *(const u32x4*)(ap + hl * 16 + 8);
    }
#pragma unroll
    for (int ni = 0; ni < 2; ++ni) {
      const unsigned short* bp = &bs[(wn * 32 + ni * 16 + l16) * BK + hl * 16];
      fb[ni].q[0] = *(const u32x4*)(bp);
      fb[ni].q[1] = *(const u32x4*)(bp + 8);
    }
#pragma unroll
    for (int mi = 0; mi < 2; ++mi)
#pragma unroll
      for (int ni = 0; ni < 2; ++ni)
        acc[mi][ni] = __builtin_amdgcn_wmma_f32_16x16x32_bf16(
            false, a[mi].v, false, fb[ni].v, (short)0, acc[mi][ni], false, false);
    __syncthreads();
  }

#pragma unroll
  for (int mi = 0; mi < 2; ++mi) {
#pragma unroll
    for (int ni = 0; ni < 2; ++ni) {
      const int col = col0 + wn * 32 + ni * 16 + l16;
      const float bb = bias ? bias[col] : 0.0f;
#pragma unroll
      for (int r = 0; r < 8; ++r) {
        const int pos = row0 + wm * 32 + mi * 16 + hl * 8 + r;
        const float v = acc[mi][ni][r];
        if (idx) {
          if (pos < cnt) {   // padding rows never touch Out
            const size_t o = (size_t)idx[pos] * DMODEL + col;
            Out[o] = Out[o] + v;
          }
        } else {
          Out[(size_t)pos * DMODEL + col] = v + bb;
        }
      }
    }
  }
}

// ---------------------------------------------------------------------------
extern "C" void kernel_launch(void* const* d_in, const int* in_sizes, int n_in,
                              void* d_out, int out_size, void* d_ws, size_t ws_size,
                              hipStream_t stream) {
  (void)in_sizes; (void)n_in; (void)out_size; (void)ws_size;
  const float* x   = (const float*)d_in[0];
  const float* gw  = (const float*)d_in[1];
  const float* w1  = (const float*)d_in[2];
  const float* w2  = (const float*)d_in[3];
  const float* w3  = (const float*)d_in[4];
  const float* sw1 = (const float*)d_in[5];
  const float* sb1 = (const float*)d_in[6];
  const float* sw2 = (const float*)d_in[7];
  const float* sb2 = (const float*)d_in[8];
  const float* sw3 = (const float*)d_in[9];
  const float* sb3 = (const float*)d_in[10];
  float* out = (float*)d_out;

  // -------- workspace layout (bytes) --------
  char* ws = (char*)d_ws;
  size_t off = 0;
  float* cw = (float*)(ws + off);        off += (size_t)T_TOK * NEXP * 4;   // 512KB
  float* stats = (float*)(ws + off);
  int* icnt    = (int*)(ws + off + 64);
  int* icntPad = (int*)(ws + off + 96);  off += 256;
  int* idxlist = (int*)(ws + off);       off += (size_t)NEXP * T_TOK * 4;   // 512KB
  unsigned short* xbf  = (unsigned short*)(ws + off); off += (size_t)T_TOK * DMODEL * 2;  // 16MB
  unsigned short* sw1t = (unsigned short*)(ws + off); off += (size_t)SHDIM * DMODEL * 2;  // 2MB
  unsigned short* sw3t = (unsigned short*)(ws + off); off += (size_t)SHDIM * DMODEL * 2;
  unsigned short* sw2t = (unsigned short*)(ws + off); off += (size_t)DMODEL * SHDIM * 2;
  unsigned short* w1t  = (unsigned short*)(ws + off); off += (size_t)NEXP * IDIM * DMODEL * 2; // 8MB
  unsigned short* w3t  = (unsigned short*)(ws + off); off += (size_t)NEXP * IDIM * DMODEL * 2;
  unsigned short* w2t  = (unsigned short*)(ws + off); off += (size_t)NEXP * DMODEL * IDIM * 2;
  unsigned short* hact = (unsigned short*)(ws + off); off += (size_t)T_TOK * SHDIM * 2;   // 64MB

  // -------- gate + routing lists + aux loss --------
  zero_stats<<<1, 32, 0, stream>>>(stats, icnt);
  gate_topk<<<T_TOK / 8, 256, 0, stream>>>(x, gw, cw, stats, icnt, idxlist);
  pad_lists<<<NEXP, 128, 0, stream>>>(icnt, icntPad, idxlist);
  aux_loss<<<1, 32, 0, stream>>>(stats, out + (size_t)T_TOK * DMODEL);

  // -------- convert-once precision prep --------
  to_bf16<<<(T_TOK * DMODEL) / 1024, 256, 0, stream>>>(x, xbf);
  transpose_to_bf16<<<dim3(SHDIM / 32, DMODEL / 32, 1), 256, 0, stream>>>(
      sw1, sw1t, DMODEL, SHDIM, 0, 0);
  transpose_to_bf16<<<dim3(SHDIM / 32, DMODEL / 32, 1), 256, 0, stream>>>(
      sw3, sw3t, DMODEL, SHDIM, 0, 0);
  transpose_to_bf16<<<dim3(DMODEL / 32, SHDIM / 32, 1), 256, 0, stream>>>(
      sw2, sw2t, SHDIM, DMODEL, 0, 0);
  transpose_to_bf16<<<dim3(IDIM / 32, DMODEL / 32, NEXP), 256, 0, stream>>>(
      w1, w1t, DMODEL, IDIM, (long)DMODEL * IDIM, (long)IDIM * DMODEL);
  transpose_to_bf16<<<dim3(IDIM / 32, DMODEL / 32, NEXP), 256, 0, stream>>>(
      w3, w3t, DMODEL, IDIM, (long)DMODEL * IDIM, (long)IDIM * DMODEL);
  transpose_to_bf16<<<dim3(DMODEL / 32, IDIM / 32, NEXP), 256, 0, stream>>>(
      w2, w2t, IDIM, DMODEL, (long)IDIM * DMODEL, (long)DMODEL * IDIM);

  // -------- shared experts (dense): out = (silu(x@sw1+b1)*(x@sw3+b3))@sw2 + b2 --------
  dual_swiglu<<<dim3(T_TOK / BM, SHDIM / BN), 256, 0, stream>>>(
      xbf, sw1t, sw3t, sb1, sb3, (const float*)nullptr, 0,
      (const int*)nullptr, (const int*)nullptr, hact, SHDIM);
  gemm_bf16_acc<<<dim3(T_TOK / BM, DMODEL / BN), 256, 0, stream>>>(
      hact, sw2t, sb2, out, SHDIM,
      (const int*)nullptr, (const int*)nullptr, (const int*)nullptr);

  // -------- routed experts (sparse gather/scatter, top-2 only) --------
  for (int e = 0; e < NEXP; ++e) {
    dual_swiglu<<<dim3(T_TOK / BM, IDIM / BN), 256, 0, stream>>>(
        xbf, w1t + (size_t)e * IDIM * DMODEL, w3t + (size_t)e * IDIM * DMODEL,
        (const float*)nullptr, (const float*)nullptr, cw, e,
        idxlist + (size_t)e * T_TOK, icntPad + e, hact, IDIM);
    gemm_bf16_acc<<<dim3(T_TOK / BM, DMODEL / BN), 256, 0, stream>>>(
        hact, w2t + (size_t)e * DMODEL * IDIM, (const float*)nullptr, out, IDIM,
        idxlist + (size_t)e * T_TOK, icnt + e, icntPad + e);
  }
}
